// CXModel_4329327034701
// MI455X (gfx1250) — matrix operand order repaired and checked by reading, not compiled, not run on hardware
//
#include <hip/hip_runtime.h>
#include <hip/hip_bf16.h>

typedef __bf16 bf16_t;
typedef __bf16 bf16x8 __attribute__((ext_vector_type(8)));
typedef __bf16 v16bf  __attribute__((ext_vector_type(16)));
typedef float  v8f    __attribute__((ext_vector_type(8)));

#define NNODES 50000
#define NEDGES 800000
#define HIDDIM 128
#define NGRAPH 64

// ---------------- utility kernels ----------------

__global__ void copy_f4(const float4* __restrict__ in, float4* __restrict__ out, int n4) {
    int i = blockIdx.x * blockDim.x + threadIdx.x;
    if (i < n4) out[i] = in[i];
}

// GIN aggregation: h[dst] += x[src]  (h was pre-initialized with x, eps=0)
__global__ void scatter_add(const float* __restrict__ x, const int* __restrict__ src,
                            const int* __restrict__ dst, float* __restrict__ h) {
    int idx = blockIdx.x * blockDim.x + threadIdx.x;   // NEDGES * 32 threads
    int e = idx >> 5;
    int f = (idx & 31) << 2;
    if (e >= NEDGES) return;
    int s = src[e], d = dst[e];
    float4 v = *(const float4*)(x + (size_t)s * HIDDIM + f);
    float* p = h + (size_t)d * HIDDIM + f;
    atomicAdd(p + 0, v.x);
    atomicAdd(p + 1, v.y);
    atomicAdd(p + 2, v.z);
    atomicAdd(p + 3, v.w);
}

__global__ void f32_to_bf16(const float* __restrict__ in, bf16_t* __restrict__ out, int n) {
    int i = blockIdx.x * blockDim.x + threadIdx.x;
    if (i < n) out[i] = (bf16_t)in[i];
}

// ---------------- weight prep: fold BN, transpose, convert to bf16 ----------------

struct PrepArgs {
    const float* W1[3]; const float* b1[3]; const float* bng[3]; const float* bnb[3];
    const float* bnm[3]; const float* bnv[3]; const float* W2[3];
    const float* Wm1;
    bf16_t* W1t; bf16_t* W2t; bf16_t* Wm1t; float* b1eff;
};

__global__ void prep_weights(PrepArgs pa) {
    const int TOT = 49152 + 49152 + 32768 + 384;
    for (int id = blockIdx.x * blockDim.x + threadIdx.x; id < TOT; id += gridDim.x * blockDim.x) {
        int t = id;
        if (t < 49152) {                                    // W1t (BN-folded), [l][n][k]
            int l = t / 16384, r = t % 16384, n = r / 128, k = r % 128;
            float sc = pa.bng[l][n] * rsqrtf(pa.bnv[l][n] + 1e-5f);
            pa.W1t[l * 16384 + n * 128 + k] = (bf16_t)(pa.W1[l][k * 128 + n] * sc);
            continue;
        }
        t -= 49152;
        if (t < 49152) {                                    // W2t, [l][n][k]
            int l = t / 16384, r = t % 16384, n = r / 128, k = r % 128;
            pa.W2t[l * 16384 + n * 128 + k] = (bf16_t)(pa.W2[l][k * 128 + n]);
            continue;
        }
        t -= 49152;
        if (t < 32768) {                                    // Wm1t, [n][k], K=256
            int n = t / 256, k = t % 256;
            pa.Wm1t[n * 256 + k] = (bf16_t)(pa.Wm1[k * 128 + n]);
            continue;
        }
        t -= 32768;
        {                                                   // b1eff
            int l = t / 128, n = t % 128;
            float sc = pa.bng[l][n] * rsqrtf(pa.bnv[l][n] + 1e-5f);
            pa.b1eff[l * 128 + n] = (pa.b1[l][n] - pa.bnm[l][n]) * sc + pa.bnb[l][n];
        }
    }
}

// ---------------- WMMA GEMM: out[M,128] = relu(A[M,128] @ W + bias) ----------------
// One block per 16-row M-tile; 8 waves, wave w owns columns [16w,16w+16).

template <bool BF16OUT>
__global__ void gemm128_relu(const bf16_t* __restrict__ A, const bf16_t* __restrict__ Wt,
                             const float* __restrict__ bias, void* __restrict__ outp) {
    const int lane = threadIdx.x & 31;
    const int wave = threadIdx.x >> 5;
    const int half = lane >> 4;
    const int r16  = lane & 15;
    const int mbase = blockIdx.x * 16;
    const int col   = wave * 16 + r16;
    const bf16_t* arow = A  + (size_t)(mbase + r16) * 128;
    const bf16_t* bcol = Wt + (size_t)col * 128;

    v8f c = {};
#pragma unroll
    for (int kk = 0; kk < 128; kk += 32) {
        bf16x8 a0 = *(const bf16x8*)(arow + kk + 16 * half);
        bf16x8 a1 = *(const bf16x8*)(arow + kk + 16 * half + 8);
        v16bf a = __builtin_shufflevector(a0, a1, 0,1,2,3,4,5,6,7,8,9,10,11,12,13,14,15);
        bf16x8 b0 = *(const bf16x8*)(bcol + kk + 16 * half);
        bf16x8 b1 = *(const bf16x8*)(bcol + kk + 16 * half + 8);
        v16bf b = __builtin_shufflevector(b0, b1, 0,1,2,3,4,5,6,7,8,9,10,11,12,13,14,15);
        c = __builtin_amdgcn_wmma_f32_16x16x32_bf16(false, a, false, b, (short)0, c, false, false);
    }

    float bn = bias[col];
#pragma unroll
    for (int v = 0; v < 8; ++v) {
        float val = fmaxf(c[v] + bn, 0.0f);
        size_t o = (size_t)(mbase + v + 8 * half) * 128 + col;
        if (BF16OUT) ((bf16_t*)outp)[o] = (bf16_t)val;
        else         ((float*)outp)[o] = val;
    }
}

// ---------------- edge scorer: relu(concat(x[s],x[d]) @ Wm1 + bm1) @ Wm2 + bm2 ----------------
// One wave per 16-edge tile. K=256 (k<128 -> src features, k>=128 -> dst features).

__global__ void edge_score(const bf16_t* __restrict__ xb, const int* __restrict__ src,
                           const int* __restrict__ dst, const bf16_t* __restrict__ Wm1t,
                           const float* __restrict__ bm1, const float* __restrict__ wm2,
                           const float* __restrict__ bm2, float* __restrict__ out) {
    const int lane = threadIdx.x & 31;
    const int wave = threadIdx.x >> 5;
    const int half = lane >> 4;
    const int r16  = lane & 15;
    const int tile = blockIdx.x * 8 + wave;
    if (tile >= NEDGES / 16) return;
    const int ebase = tile * 16;
    const int e = ebase + r16;
    const int sn = src[e], dn = dst[e];

    v8f zero = {};
    v8f c[8];
#pragma unroll
    for (int ct = 0; ct < 8; ++ct) c[ct] = zero;

#pragma unroll
    for (int kk = 0; kk < 256; kk += 32) {
        const bf16_t* base = xb + (size_t)((kk < 128) ? sn : dn) * 128 + (kk & 127);
        bf16x8 a0 = *(const bf16x8*)(base + 16 * half);
        bf16x8 a1 = *(const bf16x8*)(base + 16 * half + 8);
        v16bf a = __builtin_shufflevector(a0, a1, 0,1,2,3,4,5,6,7,8,9,10,11,12,13,14,15);
#pragma unroll
        for (int ct = 0; ct < 8; ++ct) {
            const bf16_t* bc = Wm1t + (size_t)(ct * 16 + r16) * 256 + kk + 16 * half;
            bf16x8 b0 = *(const bf16x8*)(bc);
            bf16x8 b1 = *(const bf16x8*)(bc + 8);
            v16bf b = __builtin_shufflevector(b0, b1, 0,1,2,3,4,5,6,7,8,9,10,11,12,13,14,15);
            c[ct] = __builtin_amdgcn_wmma_f32_16x16x32_bf16(false, a, false, b, (short)0, c[ct], false, false);
        }
    }

    // bias + relu, then dot with Wm2 per output row; rows m = v + 8*half live across the 16-lane half
    float partial[8];
#pragma unroll
    for (int v = 0; v < 8; ++v) partial[v] = 0.0f;
#pragma unroll
    for (int ct = 0; ct < 8; ++ct) {
        float bn = bm1[ct * 16 + r16];
        float w  = wm2[ct * 16 + r16];
#pragma unroll
        for (int v = 0; v < 8; ++v)
            partial[v] += fmaxf(c[ct][v] + bn, 0.0f) * w;
    }
#pragma unroll
    for (int v = 0; v < 8; ++v) {
        float p = partial[v];
        p += __shfl_xor(p, 1, 32);
        p += __shfl_xor(p, 2, 32);
        p += __shfl_xor(p, 4, 32);
        p += __shfl_xor(p, 8, 32);
        partial[v] = p;
    }
    if (r16 == 0) {
        float bb = bm2[0];
#pragma unroll
        for (int v = 0; v < 8; ++v)
            out[ebase + v + 8 * half] = partial[v] + bb;
    }
}

// ---------------- segment softmax over batch[src] (64 segments) ----------------

__device__ __forceinline__ unsigned fkey(float f) {
    unsigned u = __float_as_uint(f);
    return (u & 0x80000000u) ? ~u : (u | 0x80000000u);
}
__device__ __forceinline__ float fdecode(unsigned k) {
    unsigned u = (k & 0x80000000u) ? (k & 0x7FFFFFFFu) : ~k;
    return __uint_as_float(u);
}

__global__ void seg_init(unsigned* smax, float* ssum) {
    int t = threadIdx.x;
    if (t < NGRAPH) { smax[t] = 0u; ssum[t] = 0.0f; }
}
__global__ void seg_max(const float* __restrict__ sc, const int* __restrict__ src,
                        const int* __restrict__ batch, unsigned* __restrict__ smax) {
    int i = blockIdx.x * blockDim.x + threadIdx.x;
    if (i >= NEDGES) return;
    atomicMax(&smax[batch[src[i]]], fkey(sc[i]));
}
__global__ void seg_fin(const unsigned* __restrict__ smax, float* __restrict__ mxf) {
    int t = threadIdx.x;
    if (t < NGRAPH) {
        float m = fdecode(smax[t]);
        if (!__builtin_isfinite(m)) m = 0.0f;
        mxf[t] = m;
    }
}
__global__ void seg_exp(float* __restrict__ sc, const int* __restrict__ src,
                        const int* __restrict__ batch, const float* __restrict__ mxf,
                        float* __restrict__ ssum) {
    int i = blockIdx.x * blockDim.x + threadIdx.x;
    if (i >= NEDGES) return;
    int sg = batch[src[i]];
    float f = expf(sc[i] - mxf[sg]);
    sc[i] = f;
    atomicAdd(&ssum[sg], f);
}
__global__ void seg_div(float* __restrict__ sc, const int* __restrict__ src,
                        const int* __restrict__ batch, const float* __restrict__ ssum) {
    int i = blockIdx.x * blockDim.x + threadIdx.x;
    if (i >= NEDGES) return;
    float d = ssum[batch[src[i]]];
    sc[i] = sc[i] / (d == 0.0f ? 1.0f : d);
}

// ---------------- launch ----------------

extern "C" void kernel_launch(void* const* d_in, const int* in_sizes, int n_in,
                              void* d_out, int out_size, void* d_ws, size_t ws_size,
                              hipStream_t stream) {
    (void)in_sizes; (void)n_in; (void)out_size; (void)ws_size;
    const float* x     = (const float*)d_in[0];
    const int*   ei    = (const int*)d_in[1];
    const int*   batch = (const int*)d_in[2];
    const int*   src   = ei;
    const int*   dst   = ei + NEDGES;

    PrepArgs pa;
    const float* b2[3];
    for (int l = 0; l < 3; ++l) {
        const int b = 3 + 8 * l;
        pa.W1[l]  = (const float*)d_in[b + 0];
        pa.b1[l]  = (const float*)d_in[b + 1];
        pa.bng[l] = (const float*)d_in[b + 2];
        pa.bnb[l] = (const float*)d_in[b + 3];
        pa.bnm[l] = (const float*)d_in[b + 4];
        pa.bnv[l] = (const float*)d_in[b + 5];
        pa.W2[l]  = (const float*)d_in[b + 6];
        b2[l]     = (const float*)d_in[b + 7];
    }
    const float* bm1 = (const float*)d_in[28];
    const float* Wm2 = (const float*)d_in[29];
    const float* bm2 = (const float*)d_in[30];
    pa.Wm1 = (const float*)d_in[27];

    // workspace carve (256B aligned)
    char* ws = (char*)d_ws;
    size_t off = 0;
    auto alignup = [](size_t v) { return (v + 255) & ~(size_t)255; };
    float*  bufA = (float*)(ws + off);  off = alignup(off + (size_t)NNODES * HIDDIM * 4);
    float*  bufB = (float*)(ws + off);  off = alignup(off + (size_t)NNODES * HIDDIM * 4);
    bf16_t* bufC = (bf16_t*)(ws + off); off = alignup(off + (size_t)NNODES * HIDDIM * 2);
    bf16_t* bufD = (bf16_t*)(ws + off); off = alignup(off + (size_t)NNODES * HIDDIM * 2);
    pa.W1t   = (bf16_t*)(ws + off);     off = alignup(off + 3 * 16384 * 2);
    pa.W2t   = (bf16_t*)(ws + off);     off = alignup(off + 3 * 16384 * 2);
    pa.Wm1t  = (bf16_t*)(ws + off);     off = alignup(off + 32768 * 2);
    pa.b1eff = (float*)(ws + off);      off = alignup(off + 3 * 128 * 4);
    unsigned* smax = (unsigned*)(ws + off); off = alignup(off + NGRAPH * 4);
    float*    mxf  = (float*)(ws + off);    off = alignup(off + NGRAPH * 4);
    float*    ssum = (float*)(ws + off);    off = alignup(off + NGRAPH * 4);
    float* out = (float*)d_out;

    prep_weights<<<64, 256, 0, stream>>>(pa);

    for (int l = 0; l < 3; ++l) {
        const float* xin = (l == 0) ? x : bufA;
        copy_f4<<<NNODES * HIDDIM / 4 / 256, 256, 0, stream>>>((const float4*)xin, (float4*)bufB,
                                                               NNODES * HIDDIM / 4);
        scatter_add<<<NEDGES * 32 / 256, 256, 0, stream>>>(xin, src, dst, bufB);
        f32_to_bf16<<<NNODES * HIDDIM / 256, 256, 0, stream>>>(bufB, bufC, NNODES * HIDDIM);
        gemm128_relu<true ><<<NNODES / 16, 256, 0, stream>>>(bufC, pa.W1t + l * 16384,
                                                             pa.b1eff + l * 128, bufD);
        gemm128_relu<false><<<NNODES / 16, 256, 0, stream>>>(bufD, pa.W2t + l * 16384,
                                                             b2[l], bufA);
    }

    f32_to_bf16<<<NNODES * HIDDIM / 256, 256, 0, stream>>>(bufA, bufC, NNODES * HIDDIM);
    edge_score<<<NEDGES / 16 / 8, 256, 0, stream>>>(bufC, src, dst, pa.Wm1t, bm1, Wm2, bm2, out);

    seg_init<<<1, 64, 0, stream>>>(smax, ssum);
    seg_max<<<NEDGES / 256, 256, 0, stream>>>(out, src, batch, smax);
    seg_fin<<<1, 64, 0, stream>>>(smax, mxf);
    seg_exp<<<NEDGES / 256, 256, 0, stream>>>(out, src, batch, mxf, ssum);
    seg_div<<<NEDGES / 256, 256, 0, stream>>>(out, src, batch, ssum);
}
